// DeepseekV3Attention_63316407878404
// MI455X (gfx1250) — matrix operand order repaired and checked by reading, not compiled, not run on hardware
//
#include <hip/hip_runtime.h>
#include <hip/hip_bf16.h>
#include <cmath>

// ---------------------------------------------------------------------------
// DeepSeek-V3 MLA forward for MI455X (gfx1250, wave32, WMMA).
// All matmul FLOPs go through v_wmma_f32_16x16x32_bf16 (f32 accumulate).
// GEMMs: 128x128x32 tiles, 8 waves/block, double-buffered LDS; tile staging
// uses the CDNA5 async global->LDS path (ASYNCcnt) when the builtin exists.
// Attention: flash-style streaming softmax with WMMA for QK^T and PV.
// ---------------------------------------------------------------------------

typedef __bf16 bf16;
typedef __attribute__((ext_vector_type(16))) __bf16 bf16x16;
typedef __attribute__((ext_vector_type(8)))  float  f32x8;
typedef __attribute__((ext_vector_type(4)))  int    i32x4;

#if __has_builtin(__builtin_amdgcn_global_load_async_to_lds_b128) && \
    __has_builtin(__builtin_amdgcn_s_wait_asynccnt)
#define USE_ASYNC_LDS 1
#else
#define USE_ASYNC_LDS 0
#endif

#define GLOBAL_AS __attribute__((address_space(1)))
#define LDS_AS    __attribute__((address_space(3)))

// Problem constants
static constexpr int HIDDEN  = 2048;
static constexpr int NH      = 16;
static constexpr int Q_LORA  = 1536;
static constexpr int KV_LORA = 512;
static constexpr int NOPE    = 128;
static constexpr int ROPE    = 64;
static constexpr int VHDIM   = 128;
static constexpr int QHDIM   = NOPE + ROPE;      // 192
static constexpr int SEQ     = 2048;
static constexpr int BATCH   = 2;
static constexpr int BS      = BATCH * SEQ;      // 4096 rows
static constexpr int CKV_COLS = KV_LORA + ROPE;  // 576

// GEMM tiling
static constexpr int BM  = 128;
static constexpr int BN  = 128;
static constexpr int BK  = 32;
static constexpr int LDK = BK + 16;              // 48 elems -> 96B row stride (16B aligned)

// Attention tiling
static constexpr int QT = 64;                    // query tile rows
static constexpr int KT = 64;                    // key tile rows
static constexpr int PSTR = KT + 16;             // 80 elems -> 160B stride

// ---------------------------------------------------------------------------
// WMMA fragment loaders (ISA 7.12.2 layouts).
// A 16x32 bf16: lane (hgrp = lane>>4, idx = lane&15) holds row M=idx,
//   K in [8*hgrp, 8*hgrp+8) then [16+8*hgrp, 16+8*hgrp+8)  (two 16B chunks).
// B 32x16 bf16: lane holds column N=idx, K in [8*hgrp, 8*hgrp+8) then [16+8*hgrp, 16+8*hgrp+8).
// C/D 16x16 f32: register j -> row j + 8*hgrp, lane -> col idx.
// ---------------------------------------------------------------------------
__device__ __forceinline__ bf16x16 load_afrag(const bf16* rowptr, int hgrp) {
  union { bf16x16 v; uint4 u[2]; } f;
  f.u[0] = *reinterpret_cast<const uint4*>(rowptr + 8 * hgrp);
  f.u[1] = *reinterpret_cast<const uint4*>(rowptr + 16 + 8 * hgrp);
  return f.v;
}

__device__ __forceinline__ bf16x16 load_bfrag(const bf16* rowptr, int hgrp) {
  union { bf16x16 v; uint4 u[2]; } f;
  f.u[0] = *reinterpret_cast<const uint4*>(rowptr + 8 * hgrp);
  f.u[1] = *reinterpret_cast<const uint4*>(rowptr + 16 + 8 * hgrp);
  return f.v;
}

__device__ __forceinline__ f32x8 wmma_bf16(bf16x16 a, bf16x16 b, f32x8 c) {
  return __builtin_amdgcn_wmma_f32_16x16x32_bf16(
      /*neg_a=*/false, a, /*neg_b=*/false, b,
      /*c_mod=*/(short)0, c, /*reuse_a=*/false, /*reuse_b=*/false);
}

// ---------------------------------------------------------------------------
// fp32 -> bf16 conversion
// ---------------------------------------------------------------------------
__global__ void cvt_f32_bf16(const float* __restrict__ src,
                             bf16* __restrict__ dst, size_t n) {
  size_t i = (size_t)blockIdx.x * blockDim.x + threadIdx.x;
  if (i < n) dst[i] = (bf16)src[i];
}

// ---------------------------------------------------------------------------
// RMSNorm (fp32 in, bf16 out).  One block per row.
// ---------------------------------------------------------------------------
__global__ __launch_bounds__(256)
void rmsnorm_to_bf16(const float* __restrict__ X, const float* __restrict__ W,
                     bf16* __restrict__ Y, int dim, int xstride) {
  const int row = blockIdx.x;
  const float* x = X + (size_t)row * xstride;
  __shared__ float red[256];
  float s = 0.f;
  for (int c = threadIdx.x; c < dim; c += 256) { float v = x[c]; s += v * v; }
  red[threadIdx.x] = s;
  __syncthreads();
  for (int off = 128; off > 0; off >>= 1) {
    if ((int)threadIdx.x < off) red[threadIdx.x] += red[threadIdx.x + off];
    __syncthreads();
  }
  const float r = rsqrtf(red[0] / (float)dim + 1e-6f);
  bf16* y = Y + (size_t)row * dim;
  for (int c = threadIdx.x; c < dim; c += 256) y[c] = (bf16)(W[c] * x[c] * r);
}

// ---------------------------------------------------------------------------
// GEMM tile staging.
// Fast path (full tile): CDNA5 async global->LDS DMA, no VGPR round-trip.
// Slow path (edge tile): guarded load + ds_store with zero fill.
// ---------------------------------------------------------------------------
__device__ __forceinline__ void stage16_async(const bf16* g, bf16* l) {
#if USE_ASYNC_LDS
  // signature (from hipcc diagnostic): (v4i AS1*, v4i AS3*, imm offset, imm cpol)
  __builtin_amdgcn_global_load_async_to_lds_b128(
      (GLOBAL_AS i32x4*)g, (LDS_AS i32x4*)l, 0, 0);
#else
  *reinterpret_cast<uint4*>(l) = *reinterpret_cast<const uint4*>(g);
#endif
}

__device__ __forceinline__ void stage_pair_manual(const bf16* g, bf16* l, bool ok) {
  uint4 v0 = {0, 0, 0, 0}, v1 = {0, 0, 0, 0};
  if (ok) {
    const uint4* p = reinterpret_cast<const uint4*>(g);
    v0 = p[0]; v1 = p[1];
  }
  uint4* d = reinterpret_cast<uint4*>(l);
  d[0] = v0; d[1] = v1;
}

__device__ __forceinline__ void gemm_load_tile(
    const bf16* __restrict__ A, const bf16* __restrict__ B,
    int M, int N, int K, int m0, int n0, int k0,
    bf16 (*lA)[LDK], bf16 (*lB)[LDK], int tid, bool fullA, bool fullB) {
  const int row = tid >> 1;          // 0..127
  const int ko  = (tid & 1) << 4;    // 0 or 16
  {
    const bf16* g = A + (size_t)(m0 + row) * K + k0 + ko;
    bf16* l = &lA[row][ko];
    if (fullA) { stage16_async(g, l); stage16_async(g + 8, l + 8); }
    else       { stage_pair_manual(g, l, m0 + row < M); }
  }
  {
    const bf16* g = B + (size_t)(n0 + row) * K + k0 + ko;
    bf16* l = &lB[row][ko];
    if (fullB) { stage16_async(g, l); stage16_async(g + 8, l + 8); }
    else       { stage_pair_manual(g, l, n0 + row < N); }
  }
}

__device__ __forceinline__ void wait_stage() {
#if USE_ASYNC_LDS
  __builtin_amdgcn_s_wait_asynccnt(0);
#endif
}

// ---------------------------------------------------------------------------
// Tiled bf16 WMMA GEMM:  C[m,n] = sum_k A[m,k] * B[n,k]   (B = weight, N x K)
// 256 threads = 8 wave32 waves; each wave computes a 64x32 sub-tile.
// ---------------------------------------------------------------------------
__global__ __launch_bounds__(256, 2)
void gemm_bf16_nt(const bf16* __restrict__ A, const bf16* __restrict__ B,
                  float* __restrict__ C, int M, int N, int K) {
  __shared__ __align__(16) bf16 shA[2][BM][LDK];
  __shared__ __align__(16) bf16 shB[2][BN][LDK];

  const int tid  = threadIdx.x;
  const int lane = tid & 31;
  const int wave = tid >> 5;
  const int hgrp = lane >> 4;
  const int idx  = lane & 15;
  const int wr   = wave >> 2;        // 0..1 : 64-row strip
  const int wc   = wave & 3;         // 0..3 : 32-col strip
  const int m0   = blockIdx.y * BM;
  const int n0   = blockIdx.x * BN;
  const bool fullA = (m0 + BM <= M);  // uniform per block
  const bool fullB = (n0 + BN <= N);

  f32x8 acc[4][2];
  for (int r = 0; r < 4; ++r)
    for (int c = 0; c < 2; ++c)
      acc[r][c] = f32x8{0.f, 0.f, 0.f, 0.f, 0.f, 0.f, 0.f, 0.f};

  const int nK = K / BK;
  gemm_load_tile(A, B, M, N, K, m0, n0, 0, shA[0], shB[0], tid, fullA, fullB);
  wait_stage();
  __syncthreads();

  for (int kt = 0; kt < nK; ++kt) {
    const int cur = kt & 1;
    if (kt + 1 < nK)
      gemm_load_tile(A, B, M, N, K, m0, n0, (kt + 1) * BK,
                     shA[cur ^ 1], shB[cur ^ 1], tid, fullA, fullB);
    if (kt + 2 < nK) {  // lowers to global_prefetch_b8 on gfx1250
      __builtin_prefetch(A + (size_t)(m0 + (tid >> 1)) * K + (size_t)(kt + 2) * BK, 0, 0);
      __builtin_prefetch(B + (size_t)(n0 + (tid >> 1)) * K + (size_t)(kt + 2) * BK, 0, 0);
    }

    bf16x16 bfrag[2];
#pragma unroll
    for (int c = 0; c < 2; ++c)
      bfrag[c] = load_bfrag(&shB[cur][wc * 32 + c * 16 + idx][0], hgrp);
#pragma unroll
    for (int r = 0; r < 4; ++r) {
      bf16x16 afrag = load_afrag(&shA[cur][wr * 64 + r * 16 + idx][0], hgrp);
#pragma unroll
      for (int c = 0; c < 2; ++c)
        acc[r][c] = wmma_bf16(afrag, bfrag[c], acc[r][c]);
    }
    wait_stage();   // our async stores into the other buffer are complete
    __syncthreads();
  }

#pragma unroll
  for (int r = 0; r < 4; ++r)
#pragma unroll
    for (int c = 0; c < 2; ++c) {
      const int col = n0 + wc * 32 + c * 16 + idx;
      if (col >= N) continue;
#pragma unroll
      for (int j = 0; j < 8; ++j) {
        const int row = m0 + wr * 64 + r * 16 + hgrp * 8 + j;
        if (row < M) C[(size_t)row * N + col] = acc[r][c][j];
      }
    }
}

// ---------------------------------------------------------------------------
// YaRN inverse frequency (compile-time foldable constants inside)
// ---------------------------------------------------------------------------
__device__ __forceinline__ float yarn_inv_freq(int j) {
  const float dim    = 64.f;
  const float twopi  = 6.2831853071795864f;
  const float ar     = 2.f * (float)j;
  const float fe     = __powf(10000.f, -ar / dim);      // freq_extra
  const float fi     = fe / 40.f;                       // freq_inter
  const float lden   = 2.f * logf(10000.f);
  float lowf  = floorf(dim * logf(4096.f / (32.f * twopi)) / lden);
  float highf = ceilf (dim * logf(4096.f / ( 1.f * twopi)) / lden);
  lowf  = fmaxf(lowf, 0.f);
  highf = fminf(highf, dim - 1.f);
  const float denom = (highf != lowf) ? (highf - lowf) : 0.001f;
  const float ramp  = fminf(fmaxf(((float)j - lowf) / denom, 0.f), 1.f);
  const float mask  = 1.f - ramp;
  return fi * (1.f - mask) + fe * mask;
}

// ---------------------------------------------------------------------------
// Pack Q: copy nope part, apply interleaved YaRN RoPE to pe part, emit bf16
// in (b, h, s, 192) layout.  grid(S, NH, B), block 128.
// ---------------------------------------------------------------------------
__global__ __launch_bounds__(128)
void pack_query(const float* __restrict__ qsrc, const int* __restrict__ pos,
                bf16* __restrict__ Qd) {
  const int s = blockIdx.x, h = blockIdx.y, b = blockIdx.z;
  const int t = threadIdx.x;
  const size_t row = (size_t)b * SEQ + s;
  const float* src = qsrc + row * (NH * QHDIM) + (size_t)h * QHDIM;
  bf16* dst = Qd + ((size_t)(b * NH + h) * SEQ + s) * QHDIM;
  dst[t] = (bf16)src[t];  // nope (t < 128)
  if (t < 32) {
    const float p  = (float)pos[row];
    const float f  = yarn_inv_freq(t) * p;
    const float cs = __cosf(f), sn = __sinf(f);
    const float x0 = src[NOPE + 2 * t];
    const float x1 = src[NOPE + 2 * t + 1];
    dst[NOPE + t]      = (bf16)(x0 * cs - x1 * sn);
    dst[NOPE + 32 + t] = (bf16)(x1 * cs + x0 * sn);
  }
}

// ---------------------------------------------------------------------------
// Pack K/V: k_nope + roped k_pe (broadcast over heads) -> (b,h,s,192);
// V -> (b,h,s,128).  grid(S, NH, B), block 128.
// ---------------------------------------------------------------------------
__global__ __launch_bounds__(128)
void pack_kv(const float* __restrict__ kvsrc, const float* __restrict__ ckv,
             const int* __restrict__ pos,
             bf16* __restrict__ Kd, bf16* __restrict__ Vd) {
  const int s = blockIdx.x, h = blockIdx.y, b = blockIdx.z;
  const int t = threadIdx.x;
  const size_t row = (size_t)b * SEQ + s;
  const float* src = kvsrc + row * (NH * (NOPE + VHDIM)) + (size_t)h * (NOPE + VHDIM);
  bf16* kd = Kd + ((size_t)(b * NH + h) * SEQ + s) * QHDIM;
  bf16* vd = Vd + ((size_t)(b * NH + h) * SEQ + s) * VHDIM;
  kd[t] = (bf16)src[t];            // k_nope
  vd[t] = (bf16)src[NOPE + t];     // value
  if (t < 32) {
    const float* pe = ckv + row * CKV_COLS + KV_LORA;
    const float p  = (float)pos[row];
    const float f  = yarn_inv_freq(t) * p;
    const float cs = __cosf(f), sn = __sinf(f);
    const float x0 = pe[2 * t];
    const float x1 = pe[2 * t + 1];
    kd[NOPE + t]      = (bf16)(x0 * cs - x1 * sn);
    kd[NOPE + 32 + t] = (bf16)(x1 * cs + x0 * sn);
  }
}

// ---------------------------------------------------------------------------
// Flash attention, causal.  grid(S/QT, NH, B), block 128 (4 waves).
// Each wave owns a 16-row query strip; Q/K fragments loaded straight from
// global (layouts are fragment-friendly), V staged transposed in LDS,
// P round-tripped through LDS for the A-fragment swizzle.
// Output written as (b, s, NH*VHDIM) bf16 for the final o_proj GEMM.
// ---------------------------------------------------------------------------
__global__ __launch_bounds__(128)
void mla_flash(const bf16* __restrict__ Q, const bf16* __restrict__ Kb,
               const bf16* __restrict__ Vb, bf16* __restrict__ O, float scale) {
  __shared__ __align__(16) bf16 shVt[VHDIM][PSTR];  // V^T tile [128][80]
  __shared__ __align__(16) bf16 shP[QT][PSTR];      // P tile   [64][80]

  const int tid  = threadIdx.x;
  const int lane = tid & 31;
  const int wave = tid >> 5;
  const int hgrp = lane >> 4;
  const int idx  = lane & 15;
  const int qb   = blockIdx.x * QT;
  const int h    = blockIdx.y;
  const int bb   = blockIdx.z;

  const bf16* Qh = Q  + ((size_t)(bb * NH + h) * SEQ) * QHDIM;
  const bf16* Kh = Kb + ((size_t)(bb * NH + h) * SEQ) * QHDIM;
  const bf16* Vh = Vb + ((size_t)(bb * NH + h) * SEQ) * VHDIM;

  // Q fragments for this wave's 16-row strip (192 = 6 k-steps of 32)
  bf16x16 qfrag[6];
#pragma unroll
  for (int ks = 0; ks < 6; ++ks)
    qfrag[ks] = load_afrag(Qh + (size_t)(qb + wave * 16 + idx) * QHDIM + ks * 32, hgrp);

  float mrow[8], lrow[8];
  f32x8 oacc[8];
#pragma unroll
  for (int j = 0; j < 8; ++j) { mrow[j] = -1e30f; lrow[j] = 0.f; }
#pragma unroll
  for (int t = 0; t < 8; ++t)
    oacc[t] = f32x8{0.f, 0.f, 0.f, 0.f, 0.f, 0.f, 0.f, 0.f};

  for (int kt0 = 0; kt0 <= qb; kt0 += KT) {
    __syncthreads();  // protect shVt from previous iteration's readers

    // stage V^T tile: 64 key rows x 128 v-dims
    for (int it = tid; it < KT * VHDIM / 8; it += 128) {
      const int kr = it >> 4;
      const int vc = (it & 15) * 8;
      union { uint4 u; bf16 e[8]; } vv;
      vv.u = *reinterpret_cast<const uint4*>(Vh + (size_t)(kt0 + kr) * VHDIM + vc);
#pragma unroll
      for (int e = 0; e < 8; ++e) shVt[vc + e][kr] = vv.e[e];
    }

    // S = Q K^T for this wave's strip vs 64 keys (4 col-tiles x 6 k-steps)
    f32x8 sacc[4];
#pragma unroll
    for (int c = 0; c < 4; ++c)
      sacc[c] = f32x8{0.f, 0.f, 0.f, 0.f, 0.f, 0.f, 0.f, 0.f};
#pragma unroll
    for (int ks = 0; ks < 6; ++ks) {
#pragma unroll
      for (int c = 0; c < 4; ++c) {
        bf16x16 kfrag =
            load_bfrag(Kh + (size_t)(kt0 + c * 16 + idx) * QHDIM + ks * 32, hgrp);
        sacc[c] = wmma_bf16(qfrag[ks], kfrag, sacc[c]);
      }
    }

    // scale + causal mask + running row-max
    const bool diag = (kt0 == qb);
    float tmax[8];
#pragma unroll
    for (int j = 0; j < 8; ++j) tmax[j] = -1e30f;
#pragma unroll
    for (int c = 0; c < 4; ++c) {
      const int kcol = kt0 + c * 16 + idx;
#pragma unroll
      for (int j = 0; j < 8; ++j) {
        float v = sacc[c][j] * scale;
        if (diag) {
          const int qrow = qb + wave * 16 + hgrp * 8 + j;
          if (kcol > qrow) v = -1e30f;
        }
        sacc[c][j] = v;
        tmax[j] = fmaxf(tmax[j], v);
      }
    }

    // online softmax (reduce across the 16 lanes of each half-group)
#pragma unroll
    for (int j = 0; j < 8; ++j) {
      float v = tmax[j];
      v = fmaxf(v, __shfl_xor(v, 1, 32));
      v = fmaxf(v, __shfl_xor(v, 2, 32));
      v = fmaxf(v, __shfl_xor(v, 4, 32));
      v = fmaxf(v, __shfl_xor(v, 8, 32));
      const float mnew  = fmaxf(mrow[j], v);
      const float alpha = __expf(mrow[j] - mnew);
      mrow[j] = mnew;
      float ps = 0.f;
#pragma unroll
      for (int c = 0; c < 4; ++c) {
        const float p = __expf(sacc[c][j] - mnew);
        sacc[c][j] = p;
        ps += p;
      }
      ps += __shfl_xor(ps, 1, 32);
      ps += __shfl_xor(ps, 2, 32);
      ps += __shfl_xor(ps, 4, 32);
      ps += __shfl_xor(ps, 8, 32);
      lrow[j] = lrow[j] * alpha + ps;
#pragma unroll
      for (int t = 0; t < 8; ++t) oacc[t][j] *= alpha;
    }

    // write this wave's P strip (C-layout -> row-major LDS)
#pragma unroll
    for (int c = 0; c < 4; ++c)
#pragma unroll
      for (int j = 0; j < 8; ++j)
        shP[wave * 16 + hgrp * 8 + j][c * 16 + idx] = (bf16)sacc[c][j];

    __syncthreads();  // shVt staged by all waves

    // O += P V    (64 k-dim = 2 k-steps; 8 output col-tiles)
#pragma unroll
    for (int ks = 0; ks < 2; ++ks) {
      bf16x16 pfrag = load_afrag(&shP[wave * 16 + idx][ks * 32], hgrp);
#pragma unroll
      for (int t = 0; t < 8; ++t) {
        bf16x16 vfrag = load_bfrag(&shVt[t * 16 + idx][ks * 32], hgrp);
        oacc[t] = wmma_bf16(pfrag, vfrag, oacc[t]);
      }
    }
  }

  // epilogue: normalize and emit packed (b, s, NH*VHDIM)
#pragma unroll
  for (int j = 0; j < 8; ++j) {
    const float inv = 1.f / lrow[j];
    const int row = qb + wave * 16 + hgrp * 8 + j;
#pragma unroll
    for (int t = 0; t < 8; ++t) {
      const int col = t * 16 + idx;
      O[((size_t)bb * SEQ + row) * (NH * VHDIM) + (size_t)h * VHDIM + col] =
          (bf16)(oacc[t][j] * inv);
    }
  }
}

// ---------------------------------------------------------------------------
// Host-side orchestration
// ---------------------------------------------------------------------------
extern "C" void kernel_launch(void* const* d_in, const int* in_sizes, int n_in,
                              void* d_out, int out_size, void* d_ws, size_t ws_size,
                              hipStream_t stream) {
  (void)in_sizes; (void)n_in; (void)out_size; (void)ws_size;

  const float* hidden  = (const float*)d_in[0];
  const int*   posids  = (const int*)  d_in[1];
  const float* wq_a    = (const float*)d_in[2];
  const float* q_ln    = (const float*)d_in[3];
  const float* wq_b    = (const float*)d_in[4];
  const float* wkv_a   = (const float*)d_in[5];
  const float* kv_ln   = (const float*)d_in[6];
  const float* wkv_b   = (const float*)d_in[7];
  const float* wo      = (const float*)d_in[8];
  float* out = (float*)d_out;

  // workspace carve-out (256B aligned)
  size_t off = 0;
  auto alloc = [&](size_t bytes) -> char* {
    char* p = (char*)d_ws + off;
    off += (bytes + 255) & ~(size_t)255;
    return p;
  };
  bf16* h_bf    = (bf16*)alloc((size_t)BS * HIDDEN * 2);
  bf16* wqa_bf  = (bf16*)alloc((size_t)Q_LORA * HIDDEN * 2);
  bf16* wqb_bf  = (bf16*)alloc((size_t)NH * QHDIM * Q_LORA * 2);
  bf16* wkva_bf = (bf16*)alloc((size_t)CKV_COLS * HIDDEN * 2);
  bf16* wkvb_bf = (bf16*)alloc((size_t)NH * (NOPE + VHDIM) * KV_LORA * 2);
  bf16* wo_bf   = (bf16*)alloc((size_t)HIDDEN * NH * VHDIM * 2);
  float* qa_f   = (float*)alloc((size_t)BS * Q_LORA * 4);
  bf16* qan_bf  = (bf16*)alloc((size_t)BS * Q_LORA * 2);
  float* big_f  = (float*)alloc((size_t)BS * 4096 * 4);   // q_b out (3072) then kv_b out (4096)
  float* ckv_f  = (float*)alloc((size_t)BS * CKV_COLS * 4);
  bf16* ckvn_bf = (bf16*)alloc((size_t)BS * KV_LORA * 2);
  bf16* q_bf    = (bf16*)alloc((size_t)BATCH * NH * SEQ * QHDIM * 2);
  bf16* k_bf    = (bf16*)alloc((size_t)BATCH * NH * SEQ * QHDIM * 2);
  bf16* v_bf    = (bf16*)alloc((size_t)BATCH * NH * SEQ * VHDIM * 2);
  bf16* attn_bf = (bf16*)alloc((size_t)BS * NH * VHDIM * 2);

  auto cvt = [&](const float* s, bf16* d, size_t n) {
    cvt_f32_bf16<<<(unsigned)((n + 255) / 256), 256, 0, stream>>>(s, d, n);
  };
  auto gemm = [&](const bf16* A, const bf16* B, float* C, int M, int N, int K) {
    dim3 g((N + BN - 1) / BN, M / BM);
    gemm_bf16_nt<<<g, 256, 0, stream>>>(A, B, C, M, N, K);
  };

  // 1) bf16 conversions (once per launch; weights + activations)
  cvt(hidden, h_bf,    (size_t)BS * HIDDEN);
  cvt(wq_a,   wqa_bf,  (size_t)Q_LORA * HIDDEN);
  cvt(wq_b,   wqb_bf,  (size_t)NH * QHDIM * Q_LORA);
  cvt(wkv_a,  wkva_bf, (size_t)CKV_COLS * HIDDEN);
  cvt(wkv_b,  wkvb_bf, (size_t)NH * (NOPE + VHDIM) * KV_LORA);
  cvt(wo,     wo_bf,   (size_t)HIDDEN * NH * VHDIM);

  // 2) q path: q_a -> rmsnorm -> q_b -> rope/pack
  gemm(h_bf, wqa_bf, qa_f, BS, Q_LORA, HIDDEN);
  rmsnorm_to_bf16<<<BS, 256, 0, stream>>>(qa_f, q_ln, qan_bf, Q_LORA, Q_LORA);
  gemm(qan_bf, wqb_bf, big_f, BS, NH * QHDIM, Q_LORA);
  {
    dim3 g(SEQ, NH, BATCH);
    pack_query<<<g, 128, 0, stream>>>(big_f, posids, q_bf);
  }

  // 3) kv path: kv_a -> rmsnorm -> kv_b -> rope/pack (reuses big_f)
  gemm(h_bf, wkva_bf, ckv_f, BS, CKV_COLS, HIDDEN);
  rmsnorm_to_bf16<<<BS, 256, 0, stream>>>(ckv_f, kv_ln, ckvn_bf, KV_LORA, CKV_COLS);
  gemm(ckvn_bf, wkvb_bf, big_f, BS, NH * (NOPE + VHDIM), KV_LORA);
  {
    dim3 g(SEQ, NH, BATCH);
    pack_kv<<<g, 128, 0, stream>>>(big_f, ckv_f, posids, k_bf, v_bf);
  }

  // 4) attention
  const float mm = 0.1f * logf(40.f) + 1.f;            // yarn mscale
  const float scale = (mm * mm) / sqrtf((float)QHDIM); // 192^-0.5 * m^2
  {
    dim3 g(SEQ / QT, NH, BATCH);
    mla_flash<<<g, 128, 0, stream>>>(q_bf, k_bf, v_bf, attn_bf, scale);
  }

  // 5) output projection -> fp32 d_out
  gemm(attn_bf, wo_bf, out, BS, HIDDEN, NH * VHDIM);
}
